// TransformerEncoderLayer_14912126452427
// MI455X (gfx1250) — matrix operand order repaired and checked by reading, not compiled, not run on hardware
//
#include <hip/hip_runtime.h>
#include <math.h>

// Problem constants (match reference)
static constexpr int kH = 4, kC = 128, kDFF = 2048, kN = 2048, kE = 8192;
static constexpr int kHC = kH * kC; // 512

typedef __attribute__((ext_vector_type(16))) _Float16 v16h;
typedef __attribute__((ext_vector_type(8)))  float    v8f;

// ---------------------------------------------------------------------------
// WMMA fragment loaders (CDNA5 16-bit layouts, wave32)
// A 16x32 (MxK): lanes 0-15 -> M=lane, lanes16-31 -> M=lane-16 with K+8;
//   VGPR v<4: K = 2v (+8*half); VGPR v>=4: K = 16 + 2(v-4) (+8*half)
// B 32x16 (KxN) from transposed operand Bt[N][K]: lane holds column N=lane&15,
//   elements e<8 -> K = 8*(lane>>4) + e; e>=8 -> K = 16 + 8*(lane>>4) + (e-8)
// ---------------------------------------------------------------------------
static __device__ inline v16h load_a_frag(const float* __restrict__ A, int lda,
                                          int rbase, int kbase, int lane) {
  const int half = lane >> 4, r = lane & 15;
  const float* row = A + (size_t)(rbase + r) * lda;
  v16h a;
#pragma unroll
  for (int v = 0; v < 8; ++v) {
    const int kk = kbase + ((v & 4) << 2) + 2 * (v & 3) + 8 * half;
    const float2 p = *reinterpret_cast<const float2*>(row + kk);
    a[2 * v]     = (_Float16)p.x;
    a[2 * v + 1] = (_Float16)p.y;
  }
  return a;
}

static __device__ inline v16h load_b_frag(const float* __restrict__ Bt, int ldb,
                                          int nbase, int kbase, int lane) {
  const int half = lane >> 4, n = lane & 15;
  const float* row = Bt + (size_t)(nbase + n) * ldb + kbase + 8 * half;
  v16h b;
#pragma unroll
  for (int v = 0; v < 4; ++v) {
    const float4 q = *reinterpret_cast<const float4*>(row + 4 * (v & 1) + 16 * (v >> 1));
    b[4 * v]     = (_Float16)q.x;
    b[4 * v + 1] = (_Float16)q.y;
    b[4 * v + 2] = (_Float16)q.z;
    b[4 * v + 3] = (_Float16)q.w;
  }
  return b;
}

// C[m,n] = alpha * sum_k A[m,k] * Bt[n,k]  (+ bias[n])
// Block = 128 threads (4 waves). Each wave computes a 32x64 tile (8 WMMAs per
// K-step over 6 fragment loads); block covers 64 rows x 128 cols.
__global__ void gemm_wmma_f16(const float* __restrict__ A, int lda,
                              const float* __restrict__ Bt, int ldb,
                              float* __restrict__ C, int ldc,
                              const float* __restrict__ bias,
                              float alpha, int K) {
  const int lane  = threadIdx.x & 31;
  const int wave  = threadIdx.x >> 5;
  const int mbase = blockIdx.y * 64  + (wave >> 1) * 32;
  const int nbase = blockIdx.x * 128 + (wave & 1) * 64;

  v8f acc[2][4] = {};
  for (int k = 0; k < K; k += 32) {
    // Prefetch next K-tile rows toward the WGP while this step computes
    if (k + 32 < K) {
      __builtin_prefetch(A + (size_t)(mbase + (lane & 15)) * lda + (k + 32), 0, 3);
      __builtin_prefetch(A + (size_t)(mbase + 16 + (lane & 15)) * lda + (k + 32), 0, 3);
      __builtin_prefetch(Bt + (size_t)(nbase + (lane & 31)) * ldb + (k + 32), 0, 3);
      __builtin_prefetch(Bt + (size_t)(nbase + 32 + (lane & 31)) * ldb + (k + 32), 0, 3);
    }
    const v16h a0 = load_a_frag(A, lda, mbase,      k, lane);
    const v16h a1 = load_a_frag(A, lda, mbase + 16, k, lane);
    v16h b[4];
#pragma unroll
    for (int j = 0; j < 4; ++j) b[j] = load_b_frag(Bt, ldb, nbase + 16 * j, k, lane);
#pragma unroll
    for (int j = 0; j < 4; ++j) {
      acc[0][j] = __builtin_amdgcn_wmma_f32_16x16x32_f16(false, a0, false, b[j], (short)0, acc[0][j], false, false);
      acc[1][j] = __builtin_amdgcn_wmma_f32_16x16x32_f16(false, a1, false, b[j], (short)0, acc[1][j], false, false);
    }
  }

  // C/D layout: VGPR r, lane -> N = lane&15, M = r + 8*(lane>>4)
  const int half = lane >> 4, n = lane & 15;
#pragma unroll
  for (int mi = 0; mi < 2; ++mi) {
#pragma unroll
    for (int ni = 0; ni < 4; ++ni) {
      const int col = nbase + ni * 16 + n;
      const float bv = bias ? bias[col] : 0.0f;
#pragma unroll
      for (int r = 0; r < 8; ++r) {
        const int row = mbase + mi * 16 + r + 8 * half;
        C[(size_t)row * ldc + col] = acc[mi][ni][r] * alpha + bv;
      }
    }
  }
}

// ---------------------------------------------------------------------------
// Elementwise / utility kernels
// ---------------------------------------------------------------------------
__global__ void transpose_f32(const float* __restrict__ in, float* __restrict__ out,
                              int R, int Ccols) {
  const int idx = blockIdx.x * 256 + threadIdx.x;
  if (idx < R * Ccols) {
    const int r = idx / Ccols, c = idx % Ccols;
    out[(size_t)c * R + r] = in[idx];
  }
}

__global__ void gather_rows(const float* __restrict__ x, const int* __restrict__ index,
                            float* __restrict__ xj, int E, int C) {
  const int idx = blockIdx.x * 256 + threadIdx.x;
  if (idx < E * C) {
    const int e = idx / C, c = idx % C;
    xj[idx] = x[(size_t)index[e] * C + c];
  }
}

__global__ void fill_f32(float* __restrict__ p, float v, int n) {
  const int idx = blockIdx.x * 256 + threadIdx.x;
  if (idx < n) p[idx] = v;
}

__global__ void add_f32(float* __restrict__ a, const float* __restrict__ b, int n) {
  const int idx = blockIdx.x * 256 + threadIdx.x;
  if (idx < n) a[idx] += b[idx];
}

__global__ void gelu_f32(float* __restrict__ a, int n) {
  const int idx = blockIdx.x * 256 + threadIdx.x;
  if (idx < n) {
    const float v = a[idx];
    a[idx] = 0.5f * v * (1.0f + erff(v * 0.7071067811865475f));
  }
}

// Segment softmax pieces: S is [Nrows x E] per head; seg arrays are [Nrows x Nseg]
__global__ void seg_max_k(const float* __restrict__ S, const int* __restrict__ index,
                          float* __restrict__ segMax, int Nrows, int E, int Nseg) {
  const int idx = blockIdx.x * 256 + threadIdx.x;
  if (idx < Nrows * E) {
    const int n = idx / E, e = idx % E;
    atomicMax(&segMax[(size_t)n * Nseg + index[e]], S[idx]);
  }
}

__global__ void seg_expsum_k(float* __restrict__ S, const int* __restrict__ index,
                             const float* __restrict__ segMax, float* __restrict__ segSum,
                             int Nrows, int E, int Nseg) {
  const int idx = blockIdx.x * 256 + threadIdx.x;
  if (idx < Nrows * E) {
    const int n = idx / E, e = idx % E;
    const int i = index[e];
    const float v = expf(S[idx] - segMax[(size_t)n * Nseg + i]);
    S[idx] = v;
    atomicAdd(&segSum[(size_t)n * Nseg + i], v);
  }
}

__global__ void seg_norm_k(float* __restrict__ S, const int* __restrict__ index,
                           const float* __restrict__ segSum, int Nrows, int E, int Nseg) {
  const int idx = blockIdx.x * 256 + threadIdx.x;
  if (idx < Nrows * E) {
    const int n = idx / E, e = idx % E;
    S[idx] /= (segSum[(size_t)n * Nseg + index[e]] + 1e-16f);
  }
}

// Deterministic two-stage global mean/var reduction
__global__ void reduce_partial(const float* __restrict__ v, int n, float* __restrict__ part) {
  __shared__ float s1[256], s2[256];
  const int tid = threadIdx.x;
  float a = 0.0f, b = 0.0f;
  for (int j = blockIdx.x * 256 + tid; j < n; j += gridDim.x * 256) {
    const float t = v[j];
    a += t; b += t * t;
  }
  s1[tid] = a; s2[tid] = b;
  __syncthreads();
  for (int o = 128; o > 0; o >>= 1) {
    if (tid < o) { s1[tid] += s1[tid + o]; s2[tid] += s2[tid + o]; }
    __syncthreads();
  }
  if (tid == 0) { part[blockIdx.x * 2] = s1[0]; part[blockIdx.x * 2 + 1] = s2[0]; }
}

__global__ void reduce_final(const float* __restrict__ part, int nb, int n,
                             float* __restrict__ stats) {
  __shared__ float s1[256], s2[256];
  const int tid = threadIdx.x;
  float a = 0.0f, b = 0.0f;
  for (int j = tid; j < nb; j += 256) { a += part[2 * j]; b += part[2 * j + 1]; }
  s1[tid] = a; s2[tid] = b;
  __syncthreads();
  for (int o = 128; o > 0; o >>= 1) {
    if (tid < o) { s1[tid] += s1[tid + o]; s2[tid] += s2[tid + o]; }
    __syncthreads();
  }
  if (tid == 0) {
    const float mu  = s1[0] / (float)n;
    const float var = s2[0] / (float)n - mu * mu;
    stats[0] = mu;
    stats[1] = rsqrtf(var + 1e-5f);
  }
}

__global__ void ln_apply(const float* __restrict__ v, const float* __restrict__ stats,
                         const float* __restrict__ w, const float* __restrict__ b,
                         float* __restrict__ out, int n, int C) {
  const int idx = blockIdx.x * 256 + threadIdx.x;
  if (idx < n) {
    const int c = idx % C;
    out[idx] = (v[idx] - stats[0]) * stats[1] * w[c] + b[c];
  }
}

// ---------------------------------------------------------------------------
// Host-side orchestration
// ---------------------------------------------------------------------------
static inline dim3 ew_grid(int n) { return dim3((n + 255) / 256); }

extern "C" void kernel_launch(void* const* d_in, const int* in_sizes, int n_in,
                              void* d_out, int out_size, void* d_ws, size_t ws_size,
                              hipStream_t stream) {
  (void)in_sizes; (void)n_in; (void)out_size; (void)ws_size;
  const float* x    = (const float*)d_in[0];
  const int*   index= (const int*)  d_in[1];
  const float* Wq   = (const float*)d_in[2];
  const float* bq   = (const float*)d_in[3];
  const float* Wk   = (const float*)d_in[4];
  const float* bk   = (const float*)d_in[5];
  const float* Wv   = (const float*)d_in[6];
  const float* bv   = (const float*)d_in[7];
  const float* Wz   = (const float*)d_in[8];
  const float* bz   = (const float*)d_in[9];
  const float* ln_w = (const float*)d_in[10];
  const float* ln_b = (const float*)d_in[11];
  const float* W1   = (const float*)d_in[12];
  const float* b1   = (const float*)d_in[13];
  const float* W2   = (const float*)d_in[14];
  const float* b2   = (const float*)d_in[15];
  const float* W3   = (const float*)d_in[16];
  const float* b3   = (const float*)d_in[17];
  float* out = (float*)d_out;

  // Workspace carve-up (256B aligned)
  char*  base = (char*)d_ws;
  size_t off  = 0;
  auto carve = [&](size_t bytes) -> float* {
    float* p = (float*)(base + off);
    off = (off + bytes + 255) & ~(size_t)255;
    return p;
  };
  float* WqT   = carve((size_t)kC * kHC * 4);        // [512,128]
  float* WkT   = carve((size_t)kC * kHC * 4);
  float* WvT   = carve((size_t)kC * kHC * 4);
  float* WzT   = carve((size_t)kHC * kC * 4);        // [128,512]
  float* W1T   = carve((size_t)kC * kDFF * 4);       // [2048,128]
  float* W2T   = carve((size_t)kDFF * kDFF * 4);     // [2048,2048]
  float* W3T   = carve((size_t)kDFF * kC * 4);       // [128,2048]
  float* xj    = carve((size_t)kE * kC * 4);         // [8192,128]
  float* Qf    = carve((size_t)kN * kHC * 4);        // flat [N*H*C]
  float* Kf    = carve((size_t)kE * kHC * 4);        // flat [E*H*C]
  float* Vf    = carve((size_t)kE * kHC * 4);
  float* Vt    = carve((size_t)kH * kC * kE * 4);    // per-head [C,E]
  float* S     = carve((size_t)kN * kE * 4);         // per-head scratch [N,E]
  float* segMx = carve((size_t)kN * kN * 4);         // [N, Nseg]
  float* segSm = carve((size_t)kN * kN * 4);
  float* score = carve((size_t)kN * kHC * 4);        // [N, H*C]
  float* zf    = carve((size_t)kN * kC * 4);
  float* zn    = carve((size_t)kN * kC * 4);
  float* h1    = carve((size_t)kN * kDFF * 4);
  float* h2    = carve((size_t)kN * kDFF * 4);
  float* ffn   = carve((size_t)kN * kC * 4);
  float* part  = carve((size_t)1024 * 2 * 4);
  float* stats = carve((size_t)2 * 4);

  // 1) Transpose weights (fp32) so GEMM B-operand is [N,K] contiguous-K
  transpose_f32<<<ew_grid(kC * kHC), 256, 0, stream>>>(Wq, WqT, kC, kHC);
  transpose_f32<<<ew_grid(kC * kHC), 256, 0, stream>>>(Wk, WkT, kC, kHC);
  transpose_f32<<<ew_grid(kC * kHC), 256, 0, stream>>>(Wv, WvT, kC, kHC);
  transpose_f32<<<ew_grid(kHC * kC), 256, 0, stream>>>(Wz, WzT, kHC, kC);
  transpose_f32<<<ew_grid(kC * kDFF), 256, 0, stream>>>(W1, W1T, kC, kDFF);
  transpose_f32<<<ew_grid(kDFF * kDFF), 256, 0, stream>>>(W2, W2T, kDFF, kDFF);
  transpose_f32<<<ew_grid(kDFF * kC), 256, 0, stream>>>(W3, W3T, kDFF, kC);

  // 2) Edge gather x_j = x[index]
  gather_rows<<<ew_grid(kE * kC), 256, 0, stream>>>(x, index, xj, kE, kC);

  // 3) Q/K/V projections (WMMA). Grid = (Ncols/128, M/64)
  gemm_wmma_f16<<<dim3(kHC / 128, kN / 64), 128, 0, stream>>>(x,  kC, WqT, kC, Qf, kHC, bq, 1.0f, kC);
  gemm_wmma_f16<<<dim3(kHC / 128, kE / 64), 128, 0, stream>>>(xj, kC, WkT, kC, Kf, kHC, bk, 1.0f, kC);
  gemm_wmma_f16<<<dim3(kHC / 128, kE / 64), 128, 0, stream>>>(xj, kC, WvT, kC, Vf, kHC, bv, 1.0f, kC);

  // 4) Per-head V transpose: Vf + h*E*C viewed [E,C] -> Vt + h*C*E as [C,E]
  for (int h = 0; h < kH; ++h)
    transpose_f32<<<ew_grid(kE * kC), 256, 0, stream>>>(Vf + (size_t)h * kE * kC,
                                                        Vt + (size_t)h * kC * kE, kE, kC);

  // 5) Attention per head: reshape(H,N,C) on [N,H*C] is a flat reinterpret ->
  //    head h of Q is Qf + h*N*C with ld=C (same for K/V with E rows).
  const float inv_sqrt_c = 0.08838834764831845f; // 1/sqrt(128)
  for (int h = 0; h < kH; ++h) {
    const float* Qh = Qf + (size_t)h * kN * kC;
    const float* Kh = Kf + (size_t)h * kE * kC;
    const float* Vh = Vt + (size_t)h * kC * kE;

    // S = (Q K^T)/sqrt(C)  : [N,E]
    gemm_wmma_f16<<<dim3(kE / 128, kN / 64), 128, 0, stream>>>(Qh, kC, Kh, kC, S, kE,
                                                               nullptr, inv_sqrt_c, kC);
    // segment softmax over edges grouped by index, per (n)
    fill_f32<<<ew_grid(kN * kN), 256, 0, stream>>>(segMx, -INFINITY, kN * kN);
    fill_f32<<<ew_grid(kN * kN), 256, 0, stream>>>(segSm, 0.0f, kN * kN);
    seg_max_k   <<<ew_grid(kN * kE), 256, 0, stream>>>(S, index, segMx, kN, kE, kN);
    seg_expsum_k<<<ew_grid(kN * kE), 256, 0, stream>>>(S, index, segMx, segSm, kN, kE, kN);
    seg_norm_k  <<<ew_grid(kN * kE), 256, 0, stream>>>(S, index, segSm, kN, kE, kN);

    // score_h = P V : [N,C]; write at column offset h*C of [N,H*C] (handles
    // the transpose(1,0,2).reshape(N,H*C))
    gemm_wmma_f16<<<dim3(kC / 128, kN / 64), 128, 0, stream>>>(S, kE, Vh, kE,
                                                               score + (size_t)h * kC, kHC,
                                                               nullptr, 1.0f, kE);
  }

  // 6) z = score @ Wz + bz ; z += x ; graph LayerNorm
  gemm_wmma_f16<<<dim3(kC / 128, kN / 64), 128, 0, stream>>>(score, kHC, WzT, kHC, zf, kC, bz, 1.0f, kHC);
  add_f32<<<ew_grid(kN * kC), 256, 0, stream>>>(zf, x, kN * kC);
  reduce_partial<<<1024, 256, 0, stream>>>(zf, kN * kC, part);
  reduce_final<<<1, 256, 0, stream>>>(part, 1024, kN * kC, stats);
  ln_apply<<<ew_grid(kN * kC), 256, 0, stream>>>(zf, stats, ln_w, ln_b, zn, kN * kC, kC);

  // 7) FFN: gelu(zn@W1+b1) -> gelu(h1@W2+b2) -> h2@W3+b3
  gemm_wmma_f16<<<dim3(kDFF / 128, kN / 64), 128, 0, stream>>>(zn, kC, W1T, kC, h1, kDFF, b1, 1.0f, kC);
  gelu_f32<<<ew_grid(kN * kDFF), 256, 0, stream>>>(h1, kN * kDFF);
  gemm_wmma_f16<<<dim3(kDFF / 128, kN / 64), 128, 0, stream>>>(h1, kDFF, W2T, kDFF, h2, kDFF, b2, 1.0f, kDFF);
  gelu_f32<<<ew_grid(kN * kDFF), 256, 0, stream>>>(h2, kN * kDFF);
  gemm_wmma_f16<<<dim3(kC / 128, kN / 64), 128, 0, stream>>>(h2, kDFF, W3T, kDFF, ffn, kC, b3, 1.0f, kDFF);

  // 8) out = graph_layer_norm(ffn + x)
  add_f32<<<ew_grid(kN * kC), 256, 0, stream>>>(ffn, x, kN * kC);
  reduce_partial<<<1024, 256, 0, stream>>>(ffn, kN * kC, part);
  reduce_final<<<1, 256, 0, stream>>>(part, 1024, kN * kC, stats);
  ln_apply<<<ew_grid(kN * kC), 256, 0, stream>>>(ffn, stats, ln_w, ln_b, out, kN * kC, kC);
}
